// PatchAttention_69123203662042
// MI455X (gfx1250) — matrix operand order repaired and checked by reading, not compiled, not run on hardware
//
#include <hip/hip_runtime.h>

// ---------------------------------------------------------------------------
// PatchAttention for MI455X (gfx1250): bf16 WMMA GEMMs + flash attention.
// All matrix math via v_wmma_f32_16x16x32_bf16 (wave32, 16x16 tiles).
// Round 2: native bf16 converts, vectorized fragment loads, V stored
// transposed so the attention inner loop is all-contiguous b128 loads.
// ---------------------------------------------------------------------------

typedef __attribute__((ext_vector_type(16))) __bf16 v16bf;
typedef __attribute__((ext_vector_type(8)))  __bf16 v8bf;
typedef __attribute__((ext_vector_type(8)))  float  v8f;

#define DEVINL __device__ __forceinline__

constexpr int NIMG = 4;
constexpr int HW   = 72;
constexpr int PIX  = HW * HW;   // 5184
constexpr int CIN  = 64;
constexpr int EEXP = 192;
constexpr int TOT_TOK = 21504 + 24576 + 27648;   // padded tokens over 3 windows

// native f32 -> bf16 (v_cvt_pk_bf16_f32, RNE)
DEVINL __bf16 f2bf(float f) { return (__bf16)f; }

DEVINL v8f wmma_bf(v16bf a, v16bf b, v8f c) {
    return __builtin_amdgcn_wmma_f32_16x16x32_bf16(
        /*neg_a=*/false, a, /*neg_b=*/false, b,
        /*c_mod=*/(short)0, c, /*reuse_a=*/false, /*reuse_b=*/false);
}

// ---- WMMA fragment loaders (contiguous k = 16*(lane/16) + s in A and B, wave32) ----------

// A fragment, 16x32 (MxK), source row-major f32, element (m,k) = src[m*ld+k]
DEVINL v16bf a_frag_f32(const float* src, int ld, int lane) {
    const int m = lane & 15, kb = (lane >> 4) * 16;
    v16bf a;
#pragma unroll
    for (int u = 0; u < 4; ++u) {
        a[2*u+0] = f2bf(src[m*ld + kb + 2*u + 0]);
        a[2*u+1] = f2bf(src[m*ld + kb + 2*u + 1]);
        a[2*u+8] = f2bf(src[m*ld + kb + 8 + 2*u + 0]);
        a[2*u+9] = f2bf(src[m*ld + kb + 8 + 2*u + 1]);
    }
    return a;
}

// A fragment from bf16 source (global or LDS): two contiguous 16B halves
DEVINL v16bf a_frag_bf16(const __bf16* src, int ld, int lane) {
    const int m = lane & 15, kb = (lane >> 4) * 16;
    v8bf lo = *(const v8bf*)(src + m * ld + kb);
    v8bf hi = *(const v8bf*)(src + m * ld + kb + 8);
    return __builtin_shufflevector(lo, hi, 0,1,2,3,4,5,6,7,8,9,10,11,12,13,14,15);
}

// B fragment, 32x16 (KxN), source row-major, element (k,n) = src[k*ld+n]
DEVINL v16bf b_frag_f32(const float* src, int ld, int lane) {
    const int n = lane & 15, kb = (lane >> 4) * 16;
    v16bf b;
#pragma unroll
    for (int e = 0; e < 16; ++e) b[e] = f2bf(src[(kb + e) * ld + n]);
    return b;
}
// B fragment from transposed bf16 source: element (k,n) = src[n*ld + k]
// -> one contiguous 32-byte vector load per lane
DEVINL v16bf bT_frag_bf16(const __bf16* src, int ld, int lane) {
    const int n = lane & 15, kb = (lane >> 4) * 16;
    return *(const v16bf*)(src + n * ld + kb);
}

// ---- kernel 0: pixel LUT for unfold/fold permutation (exact tiling) --------
__global__ void k_lut(int* __restrict__ lut, int wv, int Kk) {
    int t = blockIdx.x * 256 + threadIdx.x;
    if (t >= PIX) return;
    int w2 = wv * wv;
    int a  = t / (Kk * w2);
    int t1 = t % (Kk * w2);
    int b  = t1 / w2;
    int t2 = t1 % w2;
    int r  = t2 / wv;
    int s  = t2 % wv;
    lut[t] = (r * Kk + a) * HW + (s * Kk + b);
}

// ---- kernel 1: expand conv1x1, expx[n,e,p] = sum_c exp_w[e,c] x[n,c,p] + b[e]
__global__ void __launch_bounds__(256) k_expand(
    const float* __restrict__ x, const float* __restrict__ ew,
    const float* __restrict__ eb, float* __restrict__ expx)
{
    const int lane = threadIdx.x & 31;
    const int wid  = blockIdx.x * 8 + (threadIdx.x >> 5);
    const int tp = wid % (PIX / 16);
    const int r1 = wid / (PIX / 16);
    const int te = r1 % (EEXP / 16);
    const int n  = r1 / (EEXP / 16);
    const int e0 = te * 16, p0 = tp * 16;
    const float* A = ew + e0 * CIN;
    const float* B = x + (long long)n * CIN * PIX + p0;
    v8f acc = {};
    acc = wmma_bf(a_frag_f32(A,      CIN, lane), b_frag_f32(B,          PIX, lane), acc);
    acc = wmma_bf(a_frag_f32(A + 32, CIN, lane), b_frag_f32(B + 32*PIX, PIX, lane), acc);
    const int col = lane & 15, mb = (lane >> 4) * 8;
    float* O = expx + ((long long)n * EEXP + e0) * PIX + p0;
#pragma unroll
    for (int r = 0; r < 8; ++r)
        O[(mb + r) * PIX + col] = acc[r] + eb[e0 + mb + r];
}

// ---- kernel 2: gather expx part into padded bf16 token sequences -----------
__global__ void k_gather(
    const float* __restrict__ expx, const int* __restrict__ lut,
    __bf16* __restrict__ useq, int widx, int wv, int L, int Lp, int ntok)
{
    int idx = blockIdx.x * 256 + threadIdx.x;
    if (idx >= ntok * CIN) return;
    int c = idx & 63;
    int g = idx >> 6;
    int m   = g % Lp;
    int seq = g / Lp;
    int w2  = wv * wv;
    int j = seq % w2;
    int n = seq / w2;
    float v = 0.0f;
    if (m < L) {
        int t = j * L + m;
        v = expx[((long long)n * EEXP + widx * CIN + c) * PIX + lut[t]];
    }
    useq[idx] = f2bf(v);
}

// ---- kernel 3: Q/K/V projections, (ntok x 64) * (64 x 64) ------------------
// Q, K stored row-major (token, c).  V stored transposed per sequence:
// Vb[(seq*64 + c)*Lp + m]  so attention V fragments are contiguous loads.
__global__ void __launch_bounds__(256) k_qkv(
    const __bf16* __restrict__ useq,
    const float* __restrict__ wq, const float* __restrict__ wk, const float* __restrict__ wvp,
    __bf16* __restrict__ Qb, __bf16* __restrict__ Kb, __bf16* __restrict__ Vb,
    int ntok, int Lp)
{
    const int lane = threadIdx.x & 31;
    int wid = blockIdx.x * 8 + (threadIdx.x >> 5);
    const int mtiles = ntok >> 4;
    if (wid >= mtiles * 12) return;
    int ntile = wid & 3;
    int rest  = wid >> 2;
    int which = rest % 3;
    int mtile = rest / 3;
    const float* W = (which == 0) ? wq : (which == 1) ? wk : wvp;
    const __bf16* A = useq + (long long)mtile * 16 * CIN;
    v8f acc = {};
    acc = wmma_bf(a_frag_bf16(A,      CIN, lane), b_frag_f32(W + ntile*16,           CIN, lane), acc);
    acc = wmma_bf(a_frag_bf16(A + 32, CIN, lane), b_frag_f32(W + 32*CIN + ntile*16,  CIN, lane), acc);
    const int col = lane & 15, mb = (lane >> 4) * 8;
    if (which == 2) {
#pragma unroll
        for (int r = 0; r < 8; ++r) {
            int tok = mtile * 16 + mb + r;
            int seq = tok / Lp;
            int m   = tok - seq * Lp;
            Vb[((long long)seq * CIN + ntile * 16 + col) * Lp + m] = f2bf(acc[r]);
        }
    } else {
        __bf16* Ob = (which == 0) ? Qb : Kb;
#pragma unroll
        for (int r = 0; r < 8; ++r)
            Ob[((long long)mtile * 16 + mb + r) * CIN + ntile * 16 + col] = f2bf(acc[r]);
    }
}

// ---- kernel 4: flash attention, one wave = one 16-row Q tile ---------------
__global__ void __launch_bounds__(128) k_attn(
    const __bf16* __restrict__ Qb, const __bf16* __restrict__ Kb, const __bf16* __restrict__ Vb,
    const int* __restrict__ lut, float* __restrict__ attn,
    int widx, int wv, int L, int Lp)
{
    __shared__ __bf16 lds_p[4 * 16 * 32];   // per-wave 16x32 P scratch
    const int lane = threadIdx.x & 31;
    const int wave = threadIdx.x >> 5;
    const int nmb  = Lp >> 6;
    const int mblk = blockIdx.x % nmb;
    const int seq  = blockIdx.x / nmb;
    const int w2 = wv * wv;
    const int j = seq % w2;
    const int n = seq / w2;
    const int m0 = mblk * 64 + wave * 16;

    const __bf16* Qt = Qb + ((long long)seq * Lp + m0) * CIN;
    const __bf16* Ks = Kb + (long long)seq * Lp * CIN;
    const __bf16* Vs = Vb + (long long)seq * CIN * Lp;   // transposed (c, m')
    __bf16* myp = lds_p + wave * 16 * 32;

    const v16bf qa0 = a_frag_bf16(Qt,      CIN, lane);
    const v16bf qa1 = a_frag_bf16(Qt + 32, CIN, lane);

    v8f o[4];
    o[0] = (v8f){}; o[1] = (v8f){}; o[2] = (v8f){}; o[3] = (v8f){};
    float rmax[8], rsum[8];
#pragma unroll
    for (int r = 0; r < 8; ++r) { rmax[r] = -3.0e38f; rsum[r] = 0.0f; }

    const int col = lane & 15, mb = (lane >> 4) * 8;
    const float scale = 0.125f;   // 1/sqrt(64)

    for (int c0 = 0; c0 < Lp; c0 += 32) {
        // S = Q * K^T : two 16-column tiles, K-dim = 64 channels (2 steps)
        v8f s0 = {}, s1 = {};
        s0 = wmma_bf(qa0, bT_frag_bf16(Ks + (long long)(c0     ) * CIN,      CIN, lane), s0);
        s0 = wmma_bf(qa1, bT_frag_bf16(Ks + (long long)(c0     ) * CIN + 32, CIN, lane), s0);
        s1 = wmma_bf(qa0, bT_frag_bf16(Ks + (long long)(c0 + 16) * CIN,      CIN, lane), s1);
        s1 = wmma_bf(qa1, bT_frag_bf16(Ks + (long long)(c0 + 16) * CIN + 32, CIN, lane), s1);

        const int g0 = c0 + col, g1 = c0 + 16 + col;
        float nm[8];
#pragma unroll
        for (int r = 0; r < 8; ++r) {
            float a = (g0 < L) ? s0[r] * scale : -1.0e30f;
            float b = (g1 < L) ? s1[r] * scale : -1.0e30f;
            s0[r] = a; s1[r] = b;
            float m = fmaxf(a, b);
            m = fmaxf(m, __shfl_xor(m, 1, 32));
            m = fmaxf(m, __shfl_xor(m, 2, 32));
            m = fmaxf(m, __shfl_xor(m, 4, 32));
            m = fmaxf(m, __shfl_xor(m, 8, 32));
            nm[r] = fmaxf(rmax[r], m);
        }
#pragma unroll
        for (int r = 0; r < 8; ++r) {
            float f  = __expf(rmax[r] - nm[r]);
            float p0 = __expf(s0[r] - nm[r]);
            float p1 = __expf(s1[r] - nm[r]);
            float ps = p0 + p1;
            ps += __shfl_xor(ps, 1, 32);
            ps += __shfl_xor(ps, 2, 32);
            ps += __shfl_xor(ps, 4, 32);
            ps += __shfl_xor(ps, 8, 32);
            rsum[r] = rsum[r] * f + ps;
            rmax[r] = nm[r];
            o[0][r] *= f; o[1][r] *= f; o[2][r] *= f; o[3][r] *= f;
            myp[(mb + r) * 32 + col]      = f2bf(p0);
            myp[(mb + r) * 32 + 16 + col] = f2bf(p1);
        }
        // same-wave LDS RAW: wait for ds stores before reading P as A-operand
        asm volatile("s_wait_dscnt 0x0" ::: "memory");
        v16bf pa = a_frag_bf16(myp, 32, lane);
#pragma unroll
        for (int ct = 0; ct < 4; ++ct) {
            v16bf vb = bT_frag_bf16(Vs + (long long)(ct * 16) * Lp + c0, Lp, lane);
            o[ct] = wmma_bf(pa, vb, o[ct]);
        }
    }

    // normalize + scatter back to pixel space
    float* abase = attn + ((long long)n * EEXP + widx * CIN) * PIX;
#pragma unroll
    for (int r = 0; r < 8; ++r) {
        int m = m0 + mb + r;
        if (m < L) {
            float inv = 1.0f / rsum[r];
            int pix = lut[j * L + m];
#pragma unroll
            for (int ct = 0; ct < 4; ++ct)
                abase[(long long)(ct * 16 + col) * PIX + pix] = o[ct][r] * inv;
        }
    }
}

// ---- kernel 5/7: per-pixel channel pooling + 2x2 convs -> boolean mask -----
__global__ void k_mask(
    const float* __restrict__ orin, const float* __restrict__ feat, int nch,
    const float* __restrict__ w1, const float* __restrict__ b1,
    const float* __restrict__ w2, const float* __restrict__ b2,
    unsigned char* __restrict__ mask)
{
    int idx = blockIdx.x * 256 + threadIdx.x;
    if (idx >= NIMG * PIX) return;
    int n = idx / PIX, pix = idx % PIX;
    const float* o = orin + (long long)n * nch * PIX + pix;
    const float* f = feat + (long long)n * nch * PIX + pix;
    float mo = -3.0e38f, so = 0.0f, mf = -3.0e38f, sf = 0.0f;
    for (int e = 0; e < nch; ++e) {
        float a = o[(long long)e * PIX]; mo = fmaxf(mo, a); so += a;
        float b = f[(long long)e * PIX]; mf = fmaxf(mf, b); sf += b;
    }
    float invn = 1.0f / (float)nch;
    so *= invn; sf *= invn;
    // cpool channel0 = max, channel1 = mean; conv1x1 with 2x2 weight (row-major O,C)
    float a0 = w1[0] * mo + w1[1] * so + b1[0];
    float a1 = w1[2] * mo + w1[3] * so + b1[1];
    float c0 = w2[0] * mf + w2[1] * sf + b2[0];
    float c1 = w2[2] * mf + w2[3] * sf + b2[1];
    // sigmoid(z) > 0.5  <=>  z > 0
    mask[idx] = (a0 * c0 + a1 * c1) > 0.0f ? 1 : 0;
}

// ---- kernel 6/8: conv1x1 GEMM with per-pixel mask-selected B operand -------
// out[n,c,p] = sum_e W[c,e] * (mask ? featT : orinT)[n,e,p] + bias[c]
__global__ void __launch_bounds__(256) k_selgemm(
    const float* __restrict__ orinT, const float* __restrict__ featT,
    const unsigned char* __restrict__ mask,
    const float* __restrict__ W, const float* __restrict__ bias,
    float* __restrict__ out, int kch)
{
    const int lane = threadIdx.x & 31;
    int wid = blockIdx.x * 8 + (threadIdx.x >> 5);
    const int tp = wid % (PIX / 16);
    const int r1 = wid / (PIX / 16);
    const int tc = r1 % 4;
    const int n  = r1 / 4;
    const int c0 = tc * 16, p0 = tp * 16;
    const int col = lane & 15, mb = (lane >> 4) * 8, kb = (lane >> 4) * 16;
    const int pix = p0 + col;
    const float* base = (mask[n * PIX + pix] ? featT : orinT);
    const float* src  = base + (long long)n * kch * PIX + pix;
    v8f acc = {};
    for (int ks = 0; ks < kch; ks += 32) {
        v16bf a = a_frag_f32(W + c0 * kch + ks, kch, lane);
        v16bf b;
#pragma unroll
        for (int e = 0; e < 16; ++e) b[e] = f2bf(src[(long long)(ks + kb + e) * PIX]);
        acc = wmma_bf(a, b, acc);
    }
    float* O = out + ((long long)n * 64 + c0) * PIX + p0;
#pragma unroll
    for (int r = 0; r < 8; ++r)
        O[(mb + r) * PIX + col] = acc[r] + bias[c0 + mb + r];
}

// ---------------------------------------------------------------------------
extern "C" void kernel_launch(void* const* d_in, const int* in_sizes, int n_in,
                              void* d_out, int out_size, void* d_ws, size_t ws_size,
                              hipStream_t stream)
{
    (void)in_sizes; (void)n_in; (void)out_size;
    const float* x      = (const float*)d_in[0];
    const float* exp_w  = (const float*)d_in[1];
    const float* exp_b  = (const float*)d_in[2];
    const float* res_w  = (const float*)d_in[3];
    const float* res_b  = (const float*)d_in[4];
    const float* fus_w  = (const float*)d_in[5];
    const float* fus_b  = (const float*)d_in[6];
    const float* wq     = (const float*)d_in[7];
    const float* wk     = (const float*)d_in[8];
    const float* wv     = (const float*)d_in[9];
    const float* mf1a_w = (const float*)d_in[10];
    const float* mf1a_b = (const float*)d_in[11];
    const float* mf1b_w = (const float*)d_in[12];
    const float* mf1b_b = (const float*)d_in[13];
    const float* mf2a_w = (const float*)d_in[14];
    const float* mf2a_b = (const float*)d_in[15];
    const float* mf2b_w = (const float*)d_in[16];
    const float* mf2b_b = (const float*)d_in[17];

    // workspace bump allocator (everything stays L2-resident, ~75 MB total)
    char* ws = (char*)d_ws;
    size_t off = 0;
    auto alloc = [&](size_t bytes) -> char* {
        char* p = ws + off;
        off = (off + bytes + 255) & ~(size_t)255;
        return p;
    };
    int*           lut  = (int*)   alloc((size_t)3 * PIX * 4);
    float*         expx = (float*) alloc((size_t)NIMG * EEXP * PIX * 4);
    float*         attn = (float*) alloc((size_t)NIMG * EEXP * PIX * 4);
    __bf16*        useq = (__bf16*)alloc((size_t)TOT_TOK * CIN * 2);
    __bf16*        Qb   = (__bf16*)alloc((size_t)TOT_TOK * CIN * 2);
    __bf16*        Kb   = (__bf16*)alloc((size_t)TOT_TOK * CIN * 2);
    __bf16*        Vb   = (__bf16*)alloc((size_t)TOT_TOK * CIN * 2);
    unsigned char* m1   = (unsigned char*)alloc((size_t)NIMG * PIX);
    float*         res  = (float*) alloc((size_t)NIMG * CIN * PIX * 4);
    unsigned char* m2   = (unsigned char*)alloc((size_t)NIMG * PIX);
    if (off > ws_size) return;   // workspace too small: bail deterministically

    struct Win { int wv, Kk, L, Lp, seqs, tokoff, ntok; };
    const Win W3[3] = {
        {2, 36, 1296, 1344,  16,     0, 21504},
        {4, 18,  324,  384,  64, 21504, 24576},
        {6, 12,  144,  192, 144, 46080, 27648},
    };

    // pixel permutation LUTs
    for (int i = 0; i < 3; ++i)
        k_lut<<<dim3((PIX + 255) / 256), 256, 0, stream>>>(lut + i * PIX, W3[i].wv, W3[i].Kk);

    // expand conv1x1 (12 x 324 x 4 tiles, 8 waves/block)
    k_expand<<<dim3((EEXP/16) * (PIX/16) * NIMG / 8), 256, 0, stream>>>(x, exp_w, exp_b, expx);

    // per window: gather -> QKV -> flash attention (+ scatter)
    for (int i = 0; i < 3; ++i) {
        const Win& W = W3[i];
        __bf16* u = useq + (size_t)W.tokoff * CIN;
        __bf16* q = Qb   + (size_t)W.tokoff * CIN;
        __bf16* k = Kb   + (size_t)W.tokoff * CIN;
        __bf16* v = Vb   + (size_t)W.tokoff * CIN;
        k_gather<<<dim3((W.ntok * CIN + 255) / 256), 256, 0, stream>>>(
            expx, lut + i * PIX, u, i, W.wv, W.L, W.Lp, W.ntok);
        k_qkv<<<dim3(((W.ntok / 16) * 12 + 7) / 8), 256, 0, stream>>>(
            u, wq, wk, wv, q, k, v, W.ntok, W.Lp);
        k_attn<<<dim3(W.seqs * (W.Lp / 64)), 128, 0, stream>>>(
            q, k, v, lut + i * PIX, attn, i, W.wv, W.L, W.Lp);
    }

    // mask fusion 1 (orin = attn, feature = expx) then res conv (64 x 192)
    k_mask<<<dim3((NIMG * PIX + 255) / 256), 256, 0, stream>>>(
        attn, expx, EEXP, mf1a_w, mf1a_b, mf1b_w, mf1b_b, m1);
    k_selgemm<<<dim3(NIMG * 4 * (PIX / 16) / 8), 256, 0, stream>>>(
        attn, expx, m1, res_w, res_b, res, EEXP);

    // mask fusion 2 (orin = res, feature = x) then fus conv (64 x 64) -> d_out
    k_mask<<<dim3((NIMG * PIX + 255) / 256), 256, 0, stream>>>(
        res, x, CIN, mf2a_w, mf2a_b, mf2b_w, mf2b_b, m2);
    k_selgemm<<<dim3(NIMG * 4 * (PIX / 16) / 8), 256, 0, stream>>>(
        res, x, m2, fus_w, fus_b, (float*)d_out, CIN);
}
